// AttentionBlock_996432413575
// MI455X (gfx1250) — matrix operand order repaired and checked by reading, not compiled, not run on hardware
//
#include <hip/hip_runtime.h>

// ---------------------------------------------------------------------------
// SAGAN-style attention block for MI455X (gfx1250, wave32).
// - Projections: packed-f32 (v_pk_fma_f32) channel matmuls + fused 2x2 pool.
// - Attention: transposed flash attention (query == lane => per-lane softmax
//   stats), normalizer via all-ones WMMA, every WMMA operand a single 32B
//   load, phi/g tiles staged via global_load_async_to_lds_b128 (ASYNCcnt
//   double-buffer pipeline), v_wmma_f32_16x16x32_f16 throughout.
// ---------------------------------------------------------------------------

typedef __attribute__((ext_vector_type(16))) _Float16 v16h;
typedef __attribute__((ext_vector_type(8)))  _Float16 v8h;
typedef __attribute__((ext_vector_type(8)))  float    v8f;
typedef __attribute__((ext_vector_type(2)))  float    v2f;

#define BB 16
#define CC 64
#define SS 4096
#define TT 1024
#define LOG2E 1.44269504088896340736f

static __device__ __forceinline__ v8f wmma_f16(v16h a, v16h b, v8f c) {
  return __builtin_amdgcn_wmma_f32_16x16x32_f16(false, a, false, b, (short)0, c,
                                                false, false);
}
// swap lane <-> lane^16 (group-of-32 ds_swizzle, xor mask 0x10)
static __device__ __forceinline__ float swz_xor16(float v) {
  return __int_as_float(
      __builtin_amdgcn_ds_swizzle(__float_as_int(v), (0x10 << 10) | 0x1f));
}
// CDNA5 async copy: 16 bytes per lane, global -> LDS, tracked by ASYNCcnt
static __device__ __forceinline__ void async_cp16(const void* src, void* lds) {
  asm volatile("global_load_async_to_lds_b128 %0, %1, off" ::"v"(
                   (unsigned)(unsigned long long)lds),
               "v"(src)
               : "memory");
}

// ---------------------------------------------------------------------------
// Kernel 1: theta = log2(e)*(w_theta @ x), two adjacent pixels per thread so
// the inner loop runs on v_pk_fma_f32 with b64 loads. Layout [B][S][32]:
// halves 0..7 real, 8..31 zero => a lane's WMMA-B operand is ONE 32B load.
// ---------------------------------------------------------------------------
__global__ __launch_bounds__(256) void proj_theta_kernel(
    const float* __restrict__ x, const float* __restrict__ w_theta,
    _Float16* __restrict__ theta) {
  int idx = blockIdx.x * 256 + threadIdx.x;   // over B*S/2
  int b = idx >> 11;
  int s2 = (idx & 2047) * 2;                  // pixel pair
  const float* xp = x + (size_t)b * CC * SS + s2;
  const v2f z2 = {0.f, 0.f};
  v2f acc[8];
#pragma unroll
  for (int o = 0; o < 8; ++o) acc[o] = z2;
#pragma unroll 4
  for (int c = 0; c < CC; ++c) {
    v2f xv = *(const v2f*)(xp + (size_t)c * SS);   // adjacent pixels, 8B load
#pragma unroll
    for (int o = 0; o < 8; ++o) acc[o] += xv * w_theta[o * CC + c];
  }
  _Float16* tp = theta + ((size_t)b * SS + s2) * 32;
#pragma unroll
  for (int o = 0; o < 8; ++o) {
    tp[o]      = (_Float16)(acc[o][0] * LOG2E);
    tp[32 + o] = (_Float16)(acc[o][1] * LOG2E);
  }
#pragma unroll
  for (int o = 8; o < 32; ++o) {
    tp[o] = (_Float16)0.f;
    tp[32 + o] = (_Float16)0.f;
  }
}

// ---------------------------------------------------------------------------
// Kernel 2: phi = maxpool2x2(w_phi @ x) -> [B][T][32] (K-padded like theta)
//           g   = maxpool2x2(w_g @ x)   -> gA [B][32][T] with each 32-key
// group kept in contiguous key order (identity, no permutation).
// The two columns of each pooling-window row are memory-adjacent ->
// accumulate them in float2 (v_pk_fma_f32), pool with packed max at the end.
// ---------------------------------------------------------------------------
__global__ __launch_bounds__(128) void proj_pool_kernel(
    const float* __restrict__ x, const float* __restrict__ w_phi,
    const float* __restrict__ w_g, _Float16* __restrict__ phi,
    _Float16* __restrict__ gA) {
  int idx = blockIdx.x * 128 + threadIdx.x;   // over B*T
  int b = idx >> 10;
  int t = idx & (TT - 1);
  int ty = t >> 5, tx = t & 31;
  const v2f z2 = {0.f, 0.f};
  float ph[8], gg[32];
#pragma unroll
  for (int o = 0; o < 8; ++o) ph[o] = -__builtin_inff();
#pragma unroll
  for (int o = 0; o < 32; ++o) gg[o] = -__builtin_inff();
  for (int pr = 0; pr < 2; ++pr) {            // two rows of the 2x2 window
    int s = (ty * 2 + pr) * 64 + tx * 2;      // columns s, s+1 adjacent
    const float* xp = x + (size_t)b * CC * SS + s;
    v2f a8[8], a32[32];
#pragma unroll
    for (int o = 0; o < 8; ++o) a8[o] = z2;
#pragma unroll
    for (int o = 0; o < 32; ++o) a32[o] = z2;
#pragma unroll 2
    for (int c = 0; c < CC; ++c) {
      v2f xv = *(const v2f*)(xp + (size_t)c * SS);
#pragma unroll
      for (int o = 0; o < 8; ++o) a8[o] += xv * w_phi[o * CC + c];
#pragma unroll
      for (int o = 0; o < 32; ++o) a32[o] += xv * w_g[o * CC + c];
    }
#pragma unroll
    for (int o = 0; o < 8; ++o)
      ph[o] = fmaxf(ph[o], fmaxf(a8[o][0], a8[o][1]));
#pragma unroll
    for (int o = 0; o < 32; ++o)
      gg[o] = fmaxf(gg[o], fmaxf(a32[o][0], a32[o][1]));
  }
  _Float16* pp = phi + (size_t)idx * 32;
#pragma unroll
  for (int o = 0; o < 8; ++o) pp[o] = (_Float16)ph[o];
#pragma unroll
  for (int o = 8; o < 32; ++o) pp[o] = (_Float16)0.f;
  // key position within its 32-group: contiguous (identity)
  int sub = (t >> 3) & 3;
  int sub2 = sub;                             // 0->0, 1->1, 2->2, 3->3
  int tperm = (t & ~31) | (sub2 << 3) | (t & 7);
#pragma unroll
  for (int o = 0; o < 32; ++o)
    gA[((size_t)b * 32 + o) * TT + tperm] = (_Float16)gg[o];
}

// ---------------------------------------------------------------------------
// Kernel 3: transposed flash attention + w_o projection + residual.
// Per 128-key tile: 8 score WMMAs, 1 ds_swizzle (max merge), vectorized P
// dump, 8 PV WMMAs + 4 ones-WMMAs. phi/g tiles (8KB each) shared by all 8
// waves -> staged once per WG via async-to-LDS, double buffered.
// ---------------------------------------------------------------------------
__global__ __launch_bounds__(256) void flash_attn_kernel(
    const _Float16* __restrict__ theta, const _Float16* __restrict__ phi,
    const _Float16* __restrict__ gA, const float* __restrict__ w_o,
    const float* __restrict__ gamma_p, const float* __restrict__ x,
    float* __restrict__ out) {
  __shared__ __align__(32) _Float16 PhL[2][128 * 32];   // 8KB x2 phi tiles
  __shared__ __align__(32) _Float16 GL[2][32 * 128];    // 8KB x2 g tiles
  __shared__ __align__(32) _Float16 Pbuf[8][16 * 128];  // 4KB per wave
  const int tid  = threadIdx.x;
  const int lane = tid & 31;
  const int wid  = tid >> 5;
  const int half = lane >> 4;
  const int l16  = lane & 15;
  const int b    = blockIdx.x;
  const int q0   = blockIdx.y * 128 + wid * 16;

  // stage one 128-key tile (phi 8KB + gA 8KB): 32B per thread per tensor
  auto stage = [&](int kt, int buf) {
    const int k0 = kt * 128;
    const _Float16* s0 = phi + ((size_t)b * TT + k0) * 32 + tid * 16;
    _Float16* d0 = &PhL[buf][tid * 16];
    async_cp16(s0, d0);
    async_cp16(s0 + 8, d0 + 8);
    const int row = tid >> 3, off = (tid & 7) * 16;
    const _Float16* s1 = gA + ((size_t)b * 32 + row) * TT + k0 + off;
    _Float16* d1 = &GL[buf][row * 128 + off];
    async_cp16(s1, d1);
    async_cp16(s1 + 8, d1 + 8);
  };

  // B operand: theta (query = lane), one 32B load thanks to padded layout
  v16h b_th =
      *(const v16h*)(theta + ((size_t)b * SS + q0 + l16) * 32 + half * 16);
  v16h a_ones;
#pragma unroll
  for (int i = 0; i < 16; ++i) a_ones[i] = (_Float16)1.0f;

  v8f accT0 = {}, accT1 = {}, accS = {};  // attn^T ch 0-15 / 16-31, normalizer
  float m = -__builtin_inff();            // per-lane (per-query) running max
  _Float16* prow = &Pbuf[wid][l16 * 128];

  stage(0, 0);
  for (int kt = 0; kt < 8; ++kt) {        // 8 tiles of 128 keys
    const int buf = kt & 1;
    if (kt < 7) {
      stage(kt + 1, buf ^ 1);
      asm volatile("s_wait_asynccnt 0x4" ::: "memory");  // tile kt landed
    } else {
      asm volatile("s_wait_asynccnt 0x0" ::: "memory");
    }
    __syncthreads();                      // tile kt visible to all waves

    // ---- scores^T: A = phi keys (single 32B LDS load), B = theta ----
    v8f sc[8];
#pragma unroll
    for (int j = 0; j < 8; ++j) {
      v16h a_ph = *(const v16h*)&PhL[buf][(j * 16 + l16) * 32 + half * 16];
      v8f z = {};
      sc[j] = wmma_f16(a_ph, b_th, z);
    }

    // ---- per-lane max over 64 key-scores + one xor16 merge ----
    float tm[8];
#pragma unroll
    for (int j = 0; j < 8; ++j) {
      float a = fmaxf(fmaxf(sc[j][0], sc[j][1]), fmaxf(sc[j][2], sc[j][3]));
      float c = fmaxf(fmaxf(sc[j][4], sc[j][5]), fmaxf(sc[j][6], sc[j][7]));
      tm[j] = fmaxf(a, c);
    }
    float vm = fmaxf(fmaxf(fmaxf(tm[0], tm[1]), fmaxf(tm[2], tm[3])),
                     fmaxf(fmaxf(tm[4], tm[5]), fmaxf(tm[6], tm[7])));
    vm = fmaxf(vm, swz_xor16(vm));
    const float mn  = fmaxf(m, vm);
    const float scl = exp2f(m - mn);      // scores are log2-domain
    m = mn;
#pragma unroll
    for (int r = 0; r < 8; ++r) {
      accT0[r] *= scl; accT1[r] *= scl; accS[r] *= scl;
    }

    // ---- P = exp2(s - m), vectorized dump (key-contiguous per lane) ----
#pragma unroll
    for (int j = 0; j < 8; ++j) {
      v8h p;
#pragma unroll
      for (int r = 0; r < 8; ++r) p[r] = (_Float16)exp2f(sc[j][r] - mn);
      *(v8h*)(prow + j * 16 + half * 8) = p;
    }
    asm volatile("s_wait_dscnt 0" ::: "memory");  // wave-local P round trip

    // ---- attn^T += g @ P^T ; normalizer += 1 @ P^T ----
#pragma unroll
    for (int c = 0; c < 4; ++c) {         // 4 chunks of 32 keys
      v16h bP = *(const v16h*)(prow + c * 32 + half * 16);
#pragma unroll
      for (int cg = 0; cg < 2; ++cg) {
        // key-contiguous gA layout => ready-made A operand, single 32B LDS load
        v16h ag =
            *(const v16h*)&GL[buf][(cg * 16 + l16) * 128 + c * 32 + half * 16];
        if (cg == 0) accT0 = wmma_f16(ag, bP, accT0);
        else         accT1 = wmma_f16(ag, bP, accT1);
      }
      accS = wmma_f16(a_ones, bP, accS);
    }
    __syncthreads();                      // done reading buf before reuse
  }

  // ---- normalize (per-lane scalar), relayout attn^T, w_o, residual ----
  const float inv = 1.0f / accS[0];
  v8h s0, s1;
#pragma unroll
  for (int r = 0; r < 8; ++r) {
    s0[r] = (_Float16)(accT0[r] * inv);
    s1[r] = (_Float16)(accT1[r] * inv);
  }
  *(v8h*)(prow + half * 8)      = s0;     // channels 0-7 / 8-15
  *(v8h*)(prow + 16 + half * 8) = s1;     // channels 16-23 / 24-31
  asm volatile("s_wait_dscnt 0" ::: "memory");
  v16h b_at = *(const v16h*)(prow + half * 16);  // B: lane=query, K=channel

  const float gma = gamma_p[0];
#pragma unroll
  for (int n0 = 0; n0 < CC; n0 += 16) {
    v16h aw;                              // A: w_o rows (oc), K=32 channels
    const float* wp = w_o + (size_t)(n0 + l16) * 32;
#pragma unroll
    for (int i = 0; i < 8; ++i) {
      aw[i]     = (_Float16)wp[half * 16 + i];
      aw[8 + i] = (_Float16)wp[half * 16 + 8 + i];
    }
    v8f z = {};
    v8f o = wmma_f16(aw, b_at, z);        // o^T: VGPR=oc row, lane=query
#pragma unroll
    for (int r = 0; r < 8; ++r) {
      const int oc = n0 + r + 8 * half;
      const size_t oi = ((size_t)b * CC + oc) * SS + q0 + l16;  // coalesced
      out[oi] = gma * o[r] + x[oi];
    }
  }
}

// ---------------------------------------------------------------------------
extern "C" void kernel_launch(void* const* d_in, const int* in_sizes, int n_in,
                              void* d_out, int out_size, void* d_ws,
                              size_t ws_size, hipStream_t stream) {
  const float* x       = (const float*)d_in[0];
  const float* w_theta = (const float*)d_in[1];
  const float* w_phi   = (const float*)d_in[2];
  const float* w_g     = (const float*)d_in[3];
  const float* w_o     = (const float*)d_in[4];
  const float* gamma   = (const float*)d_in[5];
  float* out = (float*)d_out;

  // workspace: theta 4MB | phi 1MB | gA 1MB  (all hot in 192MB L2)
  _Float16* theta = (_Float16*)d_ws;
  _Float16* phi   = theta + (size_t)BB * SS * 32;
  _Float16* gA    = phi + (size_t)BB * TT * 32;

  proj_theta_kernel<<<(BB * SS / 2) / 256, 256, 0, stream>>>(x, w_theta, theta);
  proj_pool_kernel<<<(BB * TT) / 128, 128, 0, stream>>>(x, w_phi, w_g, phi, gA);
  dim3 grid(BB, SS / 128);
  flash_attn_kernel<<<grid, 256, 0, stream>>>(theta, phi, gA, w_o, gamma, x,
                                              out);
}
